// DynamicGraphConstructor_29944511988509
// MI455X (gfx1250) — matrix operand order repaired and checked by reading, not compiled, not run on hardware
//
#include <hip/hip_runtime.h>

// ---------------------------------------------------------------------------
// DynamicGraphConstructor for MI455X (gfx1250, wave32, WMMA)
//   Phase 1 (bandwidth-bound, ~768MB @ 23.3TB/s): fused gumbel-softmax + top-3
//            one wave32 per row, online softmax + per-lane top3, shfl merge.
//   Phase 2 (matrix-bound): edge MLP as bf16 WMMA GEMM (v_wmma_f32_16x16x32_bf16)
//            [E x 512] @ [512 x 256] -> relu -> dot W2 -> sigmoid -> * top_val
// ---------------------------------------------------------------------------

typedef __attribute__((ext_vector_type(8)))  float  v8f;
typedef __attribute__((ext_vector_type(16))) __bf16 v16bf;

union FragBF {
  v16bf v;
  uint4 q[2];
};

__device__ __forceinline__ unsigned short f2bf(float f) {
  unsigned int u = __float_as_uint(f);
  unsigned int r = u + 0x7FFFu + ((u >> 16) & 1u);   // round-to-nearest-even
  return (unsigned short)(r >> 16);
}

// ---------------------------------------------------------------------------
// Kernel 0: pack W1 [512,256] f32 (row-major) into bf16 WMMA B-fragments.
// Fragment for (kstep s, ntile t): 32 lanes x 8 dwords, stored contiguously so
// the GEMM kernel issues two global_load_b128 per fragment per lane.
// Packed B order (32x16 bf16, same k order as A): lanes 0-15 hold K=0..15 (2 per VGPR), lanes 16-31 K=16..31.
// ---------------------------------------------------------------------------
__global__ __launch_bounds__(256) void pack_w1_bf16(const float* __restrict__ w1,
                                                    unsigned int* __restrict__ outp) {
  int id   = blockIdx.x * 256 + threadIdx.x;  // 65536 total
  int r    = id & 7;
  int lane = (id >> 3) & 31;
  int nt   = (id >> 8) & 15;
  int s    = id >> 12;
  int n    = nt * 16 + (lane & 15);
  int k    = s * 32 + (lane >> 4) * 16 + r * 2;
  unsigned int lo = f2bf(w1[k * 256 + n]);
  unsigned int hi = f2bf(w1[(k + 1) * 256 + n]);
  outp[id] = lo | (hi << 16);
}

// ---------------------------------------------------------------------------
// Kernel 1: fused gumbel-softmax + top-3. One wave32 per row.
// ---------------------------------------------------------------------------
__device__ __forceinline__ void online_insert(float y, int k,
                                              float& m, float& ssum,
                                              float& t0, float& t1, float& t2,
                                              int& i0, int& i1, int& i2) {
  if (y > m) { ssum = ssum * __expf(m - y) + 1.0f; m = y; }
  else       { ssum += __expf(y - m); }
  if (y > t2) {
    if (y > t0)      { t2 = t1; i2 = i1; t1 = t0; i1 = i0; t0 = y; i0 = k; }
    else if (y > t1) { t2 = t1; i2 = i1; t1 = y;  i1 = k; }
    else             { t2 = y;  i2 = k; }
  }
}

__device__ __forceinline__ void top3_insert(float y, int k,
                                            float& t0, float& t1, float& t2,
                                            int& i0, int& i1, int& i2) {
  if (y > t2) {
    if (y > t0)      { t2 = t1; i2 = i1; t1 = t0; i1 = i0; t0 = y; i0 = k; }
    else if (y > t1) { t2 = t1; i2 = i1; t1 = y;  i1 = k; }
    else             { t2 = y;  i2 = k; }
  }
}

__global__ __launch_bounds__(256) void gumbel_topk(const float* __restrict__ adj,
                                                   const float* __restrict__ gum,
                                                   const float* __restrict__ temp,
                                                   int T,
                                                   int*   __restrict__ outSrc,
                                                   int*   __restrict__ outTgt,
                                                   float* __restrict__ outW) {
  const int wv   = threadIdx.x >> 5;
  const int lane = threadIdx.x & 31;
  const int row  = blockIdx.x * 8 + wv;
  const float invT = 1.0f / temp[0];
  const float* ar = adj + (size_t)row * T;
  const float* gr = gum + (size_t)row * T;

  float m = -__builtin_inff(), ssum = 0.0f;
  float t0 = -__builtin_inff(), t1 = -__builtin_inff(), t2 = -__builtin_inff();
  int   i0 = 0, i1 = 0, i2 = 0;

  // float4 streaming: 128 elements per wave per iteration, coalesced.
  for (int k0 = lane * 4; k0 < T; k0 += 128) {
    float4 a4 = *(const float4*)(ar + k0);
    float4 g4 = *(const float4*)(gr + k0);
    online_insert((a4.x + g4.x) * invT, k0 + 0, m, ssum, t0, t1, t2, i0, i1, i2);
    online_insert((a4.y + g4.y) * invT, k0 + 1, m, ssum, t0, t1, t2, i0, i1, i2);
    online_insert((a4.z + g4.z) * invT, k0 + 2, m, ssum, t0, t1, t2, i0, i1, i2);
    online_insert((a4.w + g4.w) * invT, k0 + 3, m, ssum, t0, t1, t2, i0, i1, i2);
  }

  // wave32 butterfly merge of (max, rescaled sum) and top-3 lists
  #pragma unroll
  for (int off = 16; off >= 1; off >>= 1) {
    float om = __shfl_xor(m,    off, 32);
    float os = __shfl_xor(ssum, off, 32);
    float nm = m > om ? m : om;
    ssum = ssum * __expf(m - nm) + os * __expf(om - nm);
    m = nm;
    float o0 = __shfl_xor(t0, off, 32); int oi0 = __shfl_xor(i0, off, 32);
    float o1 = __shfl_xor(t1, off, 32); int oi1 = __shfl_xor(i1, off, 32);
    float o2 = __shfl_xor(t2, off, 32); int oi2 = __shfl_xor(i2, off, 32);
    top3_insert(o0, oi0, t0, t1, t2, i0, i1, i2);
    top3_insert(o1, oi1, t0, t1, t2, i0, i1, i2);
    top3_insert(o2, oi2, t0, t1, t2, i0, i1, i2);
  }

  if (lane == 0) {
    float inv_s = 1.0f / ssum;
    outSrc[row * 3 + 0] = row; outTgt[row * 3 + 0] = i0; outW[row * 3 + 0] = __expf(t0 - m) * inv_s;
    outSrc[row * 3 + 1] = row; outTgt[row * 3 + 1] = i1; outW[row * 3 + 1] = __expf(t1 - m) * inv_s;
    outSrc[row * 3 + 2] = row; outTgt[row * 3 + 2] = i2; outW[row * 3 + 2] = __expf(t2 - m) * inv_s;
  }
}

// ---------------------------------------------------------------------------
// Kernel 2: edge MLP with bf16 WMMA.
// Block = 32 edges x 256 outputs. 8 waves: wave w -> Mtile (w&1), Ntiles (w>>1)*4..+3.
// K = 512 in 16 steps of v_wmma_f32_16x16x32_bf16.
// LDS: 32KB buffer aliased as A-tile (32x512 bf16) then H-tile (32x256 f32).
// ---------------------------------------------------------------------------
__global__ __launch_bounds__(256) void edge_mlp(const float* __restrict__ srcF,
                                                const float* __restrict__ tgtF,
                                                const unsigned int* __restrict__ w1pack,
                                                const float* __restrict__ b1,
                                                const float* __restrict__ w2,
                                                const float* __restrict__ b2,
                                                const int* __restrict__ tgtIdx,
                                                float* __restrict__ wInOut) {
  __shared__ __align__(16) unsigned char smem[32 * 512 * 2];  // 32 KB
  unsigned short* Abf = (unsigned short*)smem;                // 32 x 512 bf16
  float*          Hs  = (float*)smem;                         // 32 x 256 f32 (aliased)

  const int tid   = threadIdx.x;
  const int eBase = blockIdx.x * 32;

  // ---- build A: concat [src_feat | tgt_feat] -> bf16 rows in LDS ----
  #pragma unroll
  for (int el = tid; el < 32 * 512; el += 256) {
    int le = el >> 9;
    int c  = el & 511;
    int e  = eBase + le;
    float f;
    if (c < 256) f = srcF[(size_t)(e / 3) * 256 + c];
    else         f = tgtF[(size_t)tgtIdx[e] * 256 + (c - 256)];
    Abf[le * 512 + c] = f2bf(f);
  }
  __syncthreads();

  // ---- WMMA GEMM: H = A @ W1 ----
  const int wv    = tid >> 5;
  const int lane  = tid & 31;
  const int mTile = wv & 1;
  const int nGrp  = wv >> 1;                  // 4 N-tiles per wave
  const int mRow  = mTile * 16 + (lane & 15);
  const int khalf = lane >> 4;

  v8f acc[4];
  #pragma unroll
  for (int t = 0; t < 4; ++t)
    #pragma unroll
    for (int v = 0; v < 8; ++v) acc[t][v] = 0.0f;

  #pragma unroll 4
  for (int s = 0; s < 16; ++s) {
    const int kbase = s * 32;
    FragBF a;
    a.q[0] = *(const uint4*)&Abf[mRow * 512 + kbase + khalf * 16];       // K 16h+0..7 of step
    a.q[1] = *(const uint4*)&Abf[mRow * 512 + kbase + khalf * 16 + 8];   // K 16h+8..15
    #pragma unroll
    for (int t = 0; t < 4; ++t) {
      const int nt = nGrp * 4 + t;
      const uint4* pb = (const uint4*)&w1pack[(((s * 16 + nt) * 32) + lane) * 8];
      FragBF b;
      b.q[0] = pb[0];
      b.q[1] = pb[1];
      acc[t] = __builtin_amdgcn_wmma_f32_16x16x32_bf16(
          /*neg_a=*/false, a.v, /*neg_b=*/false, b.v,
          /*c_mod=*/(short)0, acc[t], /*reuse_a=*/false, /*reuse_b=*/false);
    }
  }
  __syncthreads();  // everyone done reading A before H overwrites the buffer

  // ---- bias + relu, spill H to LDS (C/D layout: v->M, lane&15->N, lane>=16 -> M+8)
  {
    const int nl  = lane & 15;
    const int mOf = (lane >> 4) * 8;
    #pragma unroll
    for (int t = 0; t < 4; ++t) {
      const int n = (nGrp * 4 + t) * 16 + nl;
      const float bias = b1[n];
      #pragma unroll
      for (int v = 0; v < 8; ++v) {
        const int mm = mTile * 16 + mOf + v;
        float h = acc[t][v] + bias;
        Hs[mm * 256 + n] = h > 0.0f ? h : 0.0f;
      }
    }
  }
  __syncthreads();

  // ---- second layer: dot(H[e], w2) in groups of 8 lanes, sigmoid, * top_val ----
  {
    const int le = tid >> 3;   // 32 edges
    const int p  = tid & 7;    // 8 partials per edge
    float sum = 0.0f;
    #pragma unroll
    for (int c = 0; c < 32; ++c) {
      const int cc = p * 32 + c;
      sum += Hs[le * 256 + cc] * w2[cc];
    }
    sum += __shfl_xor(sum, 4, 32);
    sum += __shfl_xor(sum, 2, 32);
    sum += __shfl_xor(sum, 1, 32);
    if (p == 0) {
      const int e = eBase + le;
      const float z   = sum + b2[0];
      const float sig = 1.0f / (1.0f + __expf(-z));
      wInOut[e] = sig * wInOut[e];
    }
  }
}

// ---------------------------------------------------------------------------
// Launch
// ---------------------------------------------------------------------------
extern "C" void kernel_launch(void* const* d_in, const int* in_sizes, int n_in,
                              void* d_out, int out_size, void* d_ws, size_t ws_size,
                              hipStream_t stream) {
  const float* wave        = (const float*)d_in[0];   // [4,8192,256] (use batch 0)
  const float* transition  = (const float*)d_in[1];   // [4,8192,256]
  const float* target      = (const float*)d_in[2];   // [4,4096,256]
  const float* w2t_adj     = (const float*)d_in[3];   // [8192,8192]
  const float* t2t_adj     = (const float*)d_in[4];   // [8192,4096]
  const float* temperature = (const float*)d_in[5];   // [1]
  const float* g_wt        = (const float*)d_in[6];   // [8192,8192]
  const float* g_tt        = (const float*)d_in[7];   // [8192,4096]
  const float* wt_w1       = (const float*)d_in[8];   // [512,256]
  const float* wt_b1       = (const float*)d_in[9];   // [256]
  const float* wt_w2       = (const float*)d_in[10];  // [256,1]
  const float* wt_b2       = (const float*)d_in[11];  // [1]
  const float* tt_w1       = (const float*)d_in[12];
  const float* tt_b1       = (const float*)d_in[13];
  const float* tt_w2       = (const float*)d_in[14];
  const float* tt_b2       = (const float*)d_in[15];
  (void)in_sizes; (void)n_in; (void)out_size; (void)ws_size;

  float* out = (float*)d_out;
  const int E = 8192 * 3;  // 24576 edges per graph

  // output layout: [wt_edges(2E int32)] [wt_w(E f32)] [tt_edges(2E int32)] [tt_w(E f32)]
  int*   wt_src = (int*)out;
  int*   wt_tgt = wt_src + E;
  float* wt_w   = out + 2 * E;
  int*   tt_src = (int*)(out + 3 * E);
  int*   tt_tgt = tt_src + E;
  float* tt_w   = out + 5 * E;

  unsigned int* wt_pack = (unsigned int*)d_ws;        // 65536 dwords = 256 KB
  unsigned int* tt_pack = wt_pack + 65536;            // 256 KB

  // pre-pack W1 matrices into bf16 WMMA B-fragment layout (L2-resident)
  pack_w1_bf16<<<256, 256, 0, stream>>>(wt_w1, wt_pack);
  pack_w1_bf16<<<256, 256, 0, stream>>>(tt_w1, tt_pack);

  // streaming gumbel-softmax + top-3 (writes edge indices + preliminary weights)
  gumbel_topk<<<1024, 256, 0, stream>>>(w2t_adj, g_wt, temperature, 8192,
                                        wt_src, wt_tgt, wt_w);
  gumbel_topk<<<1024, 256, 0, stream>>>(t2t_adj, g_tt, temperature, 4096,
                                        tt_src, tt_tgt, tt_w);

  // WMMA edge MLP: multiplies preliminary weights in place
  edge_mlp<<<E / 32, 256, 0, stream>>>(wave, transition, wt_pack, wt_b1, wt_w2,
                                       wt_b2, wt_tgt, wt_w);
  edge_mlp<<<E / 32, 256, 0, stream>>>(transition, target, tt_pack, tt_b1, tt_w2,
                                       tt_b2, tt_tgt, tt_w);
}
